// MoeLayer_20650202759493
// MI455X (gfx1250) — matrix operand order repaired and checked by reading, not compiled, not run on hardware
//
#include <hip/hip_runtime.h>
#include <hip/hip_bf16.h>

#define T_TOK   2048
#define DDIM    1024
#define HDIM    3584
#define NEXP    8
#define MTILE   32                  // tokens per block (2 WMMA M-tiles)
#define CHUNK_H 128                 // per-iteration H chunk = 8 waves * 16
#define NCHUNK  (HDIM / CHUNK_H)    // 28
#define XP      1032                // lds_x pitch in ushorts (1024 + 8 pad)
#define HP      136                 // lds_h pitch in ushorts (128 + 8 pad)

typedef __attribute__((ext_vector_type(16))) __bf16        v16bf;
typedef __attribute__((ext_vector_type(8)))  float         v8f;
typedef __attribute__((ext_vector_type(8)))  unsigned int  v8u;

// ---------- bf16 helpers ----------
// pack(a,b): low16 = bf16(a), high16 = bf16(b) -- single v_perm_b32.
__device__ inline unsigned pk_bf16(float a, float b) {
    return __builtin_amdgcn_perm(__builtin_bit_cast(unsigned, b),   // S0 -> bytes 7:4
                                 __builtin_bit_cast(unsigned, a),   // S1 -> bytes 3:0
                                 0x07060302u);
}
__device__ inline unsigned short bf16_1(float a) {
    return (unsigned short)(__builtin_bit_cast(unsigned, a) >> 16);
}
__device__ inline v8f zero8() { v8f z = {0.f,0.f,0.f,0.f,0.f,0.f,0.f,0.f}; return z; }

// A-fragment (16x32 bf16) from LDS, contiguous K order (k = 16h + s):
//   lane<16 : M=lane,    K = kb+0..7 (v0..3), kb+8..15 (v4..7)
//   lane>=16: M=lane-16, K = kb+16..23,         kb+24..31
// caller passes p = &lds[row*pitch + kb + (hi?16:0)]
__device__ inline v16bf load_a_lds(const unsigned short* p) {
    uint4 lo = *(const uint4*)(p);
    uint4 hi = *(const uint4*)(p + 8);
    v8u v; v[0]=lo.x; v[1]=lo.y; v[2]=lo.z; v[3]=lo.w;
           v[4]=hi.x; v[5]=hi.y; v[6]=hi.z; v[7]=hi.w;
    return __builtin_bit_cast(v16bf, v);
}

// B-fragment (32x16 bf16), K contiguous in memory (w1/w3): 16 consecutive fp32.
__device__ inline v16bf load_b_row(const float* p) {
    const float4* q = (const float4*)p;
    float4 a = q[0], b = q[1], c = q[2], d = q[3];
    v8u v;
    v[0]=pk_bf16(a.x,a.y); v[1]=pk_bf16(a.z,a.w);
    v[2]=pk_bf16(b.x,b.y); v[3]=pk_bf16(b.z,b.w);
    v[4]=pk_bf16(c.x,c.y); v[5]=pk_bf16(c.z,c.w);
    v[6]=pk_bf16(d.x,d.y); v[7]=pk_bf16(d.z,d.w);
    return __builtin_bit_cast(v16bf, v);
}

// B-fragment, N contiguous (w2: B[k=h, n=d]); 16 coalesced dword loads.
__device__ inline v16bf load_b_col(const float* p) {
    v8u v;
#pragma unroll
    for (int i = 0; i < 8; i++) {
        float f0 = p[(size_t)(2*i)   * DDIM];
        float f1 = p[(size_t)(2*i+1) * DDIM];
        v[i] = pk_bf16(f0, f1);
    }
    return __builtin_bit_cast(v16bf, v);
}

__device__ inline v8f wmma_bf16(v16bf a, v16bf b, v8f c) {
    return __builtin_amdgcn_wmma_f32_16x16x32_bf16(false, a, false, b, (short)0, c, false, false);
}

// =====================================================================
// Kernel 1: gating + routing. One wave (32 lanes) per token.
// =====================================================================
__global__ __launch_bounds__(256) void moe_gate(
    const float* __restrict__ x, const float* __restrict__ gw,
    const int* __restrict__ drafting,
    int* counts, int* pos, float* gates, int* etok)
{
    __shared__ float s_gw[NEXP * DDIM];   // 32 KB
    const int tid = threadIdx.x;
#pragma unroll
    for (int i = 0; i < (NEXP * DDIM) / 256; i++)
        s_gw[tid + 256 * i] = gw[tid + 256 * i];
    __syncthreads();

    const int wave = tid >> 5, lane = tid & 31;
    const int t = blockIdx.x * 8 + wave;
    const float* xr = x + (size_t)t * DDIM;

    float acc[NEXP];
#pragma unroll
    for (int e = 0; e < NEXP; e++) acc[e] = 0.f;
    for (int i = lane; i < DDIM; i += 32) {
        float xv = xr[i];
#pragma unroll
        for (int e = 0; e < NEXP; e++)
            acc[e] += xv * s_gw[e * DDIM + i];
    }
#pragma unroll
    for (int e = 0; e < NEXP; e++)
#pragma unroll
        for (int off = 16; off >= 1; off >>= 1)
            acc[e] += __shfl_xor(acc[e], off, 32);

    if (lane == 0) {
        float best = -3.4e38f, sec = -3.4e38f; int be = 0, se = 0;
#pragma unroll
        for (int e = 0; e < NEXP; e++) {
            float v = acc[e];
            if (v > best)      { sec = best; se = be; best = v; be = e; }
            else if (v > sec)  { sec = v; se = e; }
        }
        const int k = (drafting[0] != 0) ? 1 : 2;
        float w0, w1v;
        if (k == 1) { w0 = 1.f; w1v = 0.f; }
        else        { w0 = 1.f / (1.f + __expf(sec - best)); w1v = 1.f - w0; }

        int s0 = atomicAdd(&counts[be], 1);
        etok[be * T_TOK + s0] = t;
        int p0 = be * T_TOK + s0;
        int p1 = p0;
        if (k == 2) {
            int s1 = atomicAdd(&counts[se], 1);
            etok[se * T_TOK + s1] = t;
            p1 = se * T_TOK + s1;
        }
        pos[2*t]     = p0;   gates[2*t]     = w0;
        pos[2*t + 1] = p1;   gates[2*t + 1] = w1v;
    }
}

// =====================================================================
// Kernel 2: 8-element exclusive prefix sum -> packed contrib rows.
// =====================================================================
__global__ void moe_offsets(const int* __restrict__ counts, int* __restrict__ offsets)
{
    if (threadIdx.x == 0) {
        int acc = 0;
#pragma unroll
        for (int e = 0; e < NEXP; e++) { offsets[e] = acc; acc += counts[e]; }
    }
}

// =====================================================================
// Kernel 3: fused SwiGLU FFN per (expert, 32-token tile). 8 waves.
// Every streamed weight B-fragment is reused by 2 M-tiles.
// =====================================================================
__global__ __launch_bounds__(256) void moe_ffn(
    const float* __restrict__ x,  const float* __restrict__ w1,
    const float* __restrict__ w2, const float* __restrict__ w3,
    const int* __restrict__ counts, const int* __restrict__ offsets,
    const int* __restrict__ etok, float* __restrict__ contrib)
{
    __shared__ unsigned short lds_x[MTILE * XP];   // 66.0 KB bf16 X tile
    __shared__ unsigned short lds_h[MTILE * HP];   //  8.7 KB bf16 h tile
    __shared__ int s_tok[MTILE];

    const int e     = blockIdx.x >> 6;
    const int mt    = blockIdx.x & 63;
    const int cnt   = counts[e];
    const int rbase = mt * MTILE;
    if (rbase >= cnt) return;                      // block-uniform exit
    const int nrows = min(MTILE, cnt - rbase);

    const int tid   = threadIdx.x;
    const int wave  = tid >> 5;
    const int lane  = tid & 31;
    const int l15   = lane & 15;
    const bool hiH  = lane >= 16;
    const int asel  = hiH ? 16 : 0;   // A-fragment K sub-offset
    const int bsel  = hiH ? 16 : 0;   // B-fragment K sub-offset

    if (tid < MTILE) {
        int s = (tid < nrows) ? tid : 0;           // pad rows: any valid token
        s_tok[tid] = etok[e * T_TOK + rbase + s];
    }
    __syncthreads();

    // ---- stage X tile as bf16 into LDS (row-major, padded pitch) ----
    {
        const int row = tid >> 3;                  // 32 rows x 8 threads
        const int c0  = (tid & 7) * 128;           // 128 floats per thread
        const float4* xr = (const float4*)(x + (size_t)s_tok[row] * DDIM + c0);
        unsigned int* dst = (unsigned int*)&lds_x[row * XP + c0];
#pragma unroll
        for (int i = 0; i < 32; i++) {
            float4 f = xr[i];
            dst[2*i]   = pk_bf16(f.x, f.y);
            dst[2*i+1] = pk_bf16(f.z, f.w);
        }
    }
    __syncthreads();

    const size_t wbase = (size_t)e * HDIM * DDIM;

    v8f acc_out[16];                               // wave's 32x128 fp32 tile
#pragma unroll
    for (int i = 0; i < 16; i++) acc_out[i] = zero8();

    for (int ch = 0; ch < NCHUNK; ch++) {
        const int h0 = ch * CHUNK_H + wave * 16;   // this wave's 16 H cols

        if (ch + 1 < NCHUNK)                       // hint next weight chunk
            __builtin_prefetch(w1 + wbase + (size_t)(h0 + CHUNK_H + l15) * DDIM, 0, 1);

        // -------- phase 1: a1 = X@w1^T, a3 = X@w3^T  (K = 1024) --------
        v8f a1[2] = { zero8(), zero8() };
        v8f a3[2] = { zero8(), zero8() };
        const size_t rowoff = wbase + (size_t)(h0 + l15) * DDIM + bsel;
#pragma unroll 2
        for (int ks = 0; ks < 32; ks++) {
            const int kb = ks * 32;
            v16bf A0 = load_a_lds(&lds_x[(0  + l15) * XP + kb + asel]);
            v16bf A1 = load_a_lds(&lds_x[(16 + l15) * XP + kb + asel]);
            v16bf B1 = load_b_row(w1 + rowoff + kb);
            a1[0] = wmma_bf16(A0, B1, a1[0]);
            a1[1] = wmma_bf16(A1, B1, a1[1]);
            v16bf B3 = load_b_row(w3 + rowoff + kb);
            a3[0] = wmma_bf16(A0, B3, a3[0]);
            a3[1] = wmma_bf16(A1, B3, a3[1]);
        }

        // -------- SiLU(a1)*a3 -> bf16 h tile in LDS --------
        {
            const int hcol = wave * 16 + l15;
#pragma unroll
            for (int m = 0; m < 2; m++)
#pragma unroll
                for (int r = 0; r < 8; r++) {
                    const int row = m * 16 + r + (hiH ? 8 : 0);
                    float g  = a1[m][r];
                    float hv = (g / (1.f + __expf(-g))) * a3[m][r];
                    lds_h[row * HP + hcol] = bf16_1(hv);
                }
        }
        __syncthreads();

        // -------- phase 2: out += h @ w2  (K = 128 this chunk) --------
#pragma unroll 2
        for (int ks = 0; ks < 4; ks++) {
            const int kb = ks * 32;
            v16bf A0 = load_a_lds(&lds_h[(0  + l15) * HP + kb + asel]);
            v16bf A1 = load_a_lds(&lds_h[(16 + l15) * HP + kb + asel]);
            const size_t kglob = (size_t)(ch * CHUNK_H + kb + bsel);
#pragma unroll
            for (int nt = 0; nt < 8; nt++) {
                const int dcol = wave * 128 + nt * 16 + l15;
                v16bf B = load_b_col(w2 + wbase + kglob * DDIM + dcol);
                acc_out[nt*2+0] = wmma_bf16(A0, B, acc_out[nt*2+0]);
                acc_out[nt*2+1] = wmma_bf16(A1, B, acc_out[nt*2+1]);
            }
        }
        __syncthreads();                           // lds_h reused next chunk
    }

    // -------- epilogue: write packed contribution rows (no atomics) --------
    const int obase = offsets[e] + rbase;
#pragma unroll
    for (int nt = 0; nt < 8; nt++) {
        const int dcol = wave * 128 + nt * 16 + l15;
#pragma unroll
        for (int m = 0; m < 2; m++)
#pragma unroll
            for (int r = 0; r < 8; r++) {
                const int row = m * 16 + r + (hiH ? 8 : 0);
                if (row < nrows)
                    contrib[(size_t)(obase + row) * DDIM + dcol] = acc_out[nt*2+m][r];
            }
    }
}

// =====================================================================
// Kernel 4: out[t] = g0*contrib[row(p0)] + g1*contrib[row(p1)]
// =====================================================================
__global__ __launch_bounds__(256) void moe_combine(
    const int* __restrict__ pos, const float* __restrict__ gates,
    const int* __restrict__ offsets, const float* __restrict__ contrib,
    float* __restrict__ out)
{
    const int t = blockIdx.x;
    const int d = threadIdx.x * 4;
    const int p0 = pos[2*t], p1 = pos[2*t + 1];
    const float g0 = gates[2*t], g1 = gates[2*t + 1];
    const int r0 = offsets[p0 >> 11] + (p0 & (T_TOK - 1));
    const int r1 = offsets[p1 >> 11] + (p1 & (T_TOK - 1));
    float4 c0 = *(const float4*)(contrib + (size_t)r0 * DDIM + d);
    float4 c1 = *(const float4*)(contrib + (size_t)r1 * DDIM + d);
    float4 o;
    o.x = g0 * c0.x + g1 * c1.x;
    o.y = g0 * c0.y + g1 * c1.y;
    o.z = g0 * c0.z + g1 * c1.z;
    o.w = g0 * c0.w + g1 * c1.w;
    *(float4*)(out + (size_t)t * DDIM + d) = o;
}

// =====================================================================
extern "C" void kernel_launch(void* const* d_in, const int* in_sizes, int n_in,
                              void* d_out, int out_size, void* d_ws, size_t ws_size,
                              hipStream_t stream)
{
    const float* x        = (const float*)d_in[0];
    const float* gw       = (const float*)d_in[1];
    const float* w1       = (const float*)d_in[2];
    const float* w2       = (const float*)d_in[3];
    const float* w3       = (const float*)d_in[4];
    const int*   drafting = (const int*)d_in[5];
    float*       out      = (float*)d_out;

    // workspace layout (all 16B-aligned where vector-accessed)
    int*   counts  = (int*)d_ws;                 // E
    int*   offsets = counts + NEXP;              // E
    int*   pos     = offsets + NEXP;             // 2T
    float* gates   = (float*)(pos + 2 * T_TOK);  // 2T
    int*   etok    = (int*)(gates + 2 * T_TOK);  // E*T
    float* contrib = (float*)(etok + NEXP * T_TOK); // 2T * D

    hipMemsetAsync(counts, 0, NEXP * sizeof(int), stream);
    moe_gate   <<<T_TOK / 8, 256, 0, stream>>>(x, gw, drafting, counts, pos, gates, etok);
    moe_offsets<<<1, 32, 0, stream>>>(counts, offsets);
    moe_ffn    <<<NEXP * (T_TOK / MTILE), 256, 0, stream>>>(x, w1, w2, w3, counts, offsets, etok, contrib);
    moe_combine<<<T_TOK, 256, 0, stream>>>(pos, gates, offsets, contrib, out);
}
